// GraphBiasedAttention_7705171329292
// MI455X (gfx1250) — matrix operand order repaired and checked by reading, not compiled, not run on hardware
//
#include <hip/hip_runtime.h>
#include <stdint.h>
#include <stddef.h>

#define B_  8
#define N_  1024
#define D_  512
#define H_  8
#define DH_ 64
#define GH_ 32

#define GT_  512   // gate table cells per axis (513 grid points)
#define BT_  1024  // bias table cells (1025 grid points)

typedef __attribute__((ext_vector_type(16))) __bf16        v16bf;
typedef __attribute__((ext_vector_type(8)))  float         v8f;
typedef __attribute__((ext_vector_type(4)))  unsigned int  u32x4;

#if __has_builtin(__builtin_amdgcn_global_load_async_to_lds_b128) && \
    __has_builtin(__builtin_amdgcn_s_wait_asynccnt)
#define ASYNC_LDS 1
typedef int v4i_vs __attribute__((vector_size(16)));
typedef __attribute__((address_space(1))) v4i_vs* gvp_t;   // global v4i*
typedef __attribute__((address_space(3))) v4i_vs* lvp_t;   // LDS v4i*
static __device__ __forceinline__ void async_copy16(const void* g, void* l) {
  __builtin_amdgcn_global_load_async_to_lds_b128((gvp_t)g, (lvp_t)l, 0, 0);
}
#endif

struct Frag32B { u32x4 lo, hi; };

static __device__ __forceinline__ unsigned short f2bf(float f) {
  unsigned int u = __builtin_bit_cast(unsigned int, f);
  u += 0x7FFFu + ((u >> 16) & 1u);           // round-to-nearest-even
  return (unsigned short)(u >> 16);
}

// A fragment (16x32 bf16): lane holds row (lane&15); elems [0..15]=K (half*16 ..)
// contiguous, matching the B fragment k order.
static __device__ __forceinline__ v16bf load_a_frag(const unsigned short* base,
                                                    int stride, int lane) {
  int r = lane & 15, hh = lane >> 4;
  Frag32B t;
  t.lo = *(const u32x4*)(base + (size_t)r * stride + hh * 16);
  t.hi = *(const u32x4*)(base + (size_t)r * stride + hh * 16 + 8);
  return __builtin_bit_cast(v16bf, t);
}

// B fragment (32x16 bf16): lane holds column (lane&15); 16 contiguous K at
// (lane>>4)*16. Source is row-major [Ncols][K] (weights / key rows / V^T).
static __device__ __forceinline__ v16bf load_b_frag(const unsigned short* base,
                                                    int stride, int lane) {
  int n = lane & 15, hh = lane >> 4;
  const unsigned short* p = base + (size_t)n * stride + hh * 16;
  Frag32B t;
  t.lo = *(const u32x4*)(p);
  t.hi = *(const u32x4*)(p + 8);
  return __builtin_bit_cast(v16bf, t);
}

static __device__ __forceinline__ v8f wmma_bf16(v16bf a, v16bf b, v8f c) {
  return __builtin_amdgcn_wmma_f32_16x16x32_bf16(false, a, false, b,
                                                 (short)0, c, false, false);
}

// exact gate MLP (used for table build + out-of-range fallback)
static __device__ __forceinline__ float gate_exact(
    float ae, float ap, const float* __restrict__ W1, const float* __restrict__ b1,
    const float* __restrict__ W2, const float* __restrict__ b2) {
  float z = b2[0];
#pragma unroll
  for (int j = 0; j < GH_; ++j) {
    float t = W1[2 * j] * ae + W1[2 * j + 1] * ap + b1[j];
    float gelu = 0.5f * t * (1.0f + erff(t * 0.70710678118654752f));
    z += W2[j] * gelu;
  }
  return 1.0f / (1.0f + __expf(-z));
}

static __device__ __forceinline__ float softplus_exact(float u) {
  return (u > 20.0f) ? u : __logf(1.0f + __expf(u));
}

// ---------------------------------------------------------------- f32 -> bf16
__global__ void __launch_bounds__(256) cvt_kernel(const float* __restrict__ src,
                                                  unsigned short* __restrict__ dst,
                                                  int n) {
  int i = blockIdx.x * 256 + threadIdx.x;
  if (i < n) dst[i] = f2bf(src[i]);
}

// ----------------- table builders: 2D gate table, per-head 1D softplus tables
__global__ void __launch_bounds__(256) build_gate_table(
    const float* __restrict__ W1, const float* __restrict__ b1,
    const float* __restrict__ W2, const float* __restrict__ b2,
    float* __restrict__ Tg) {
  int i = blockIdx.x * 256 + threadIdx.x;
  if (i >= (GT_ + 1) * (GT_ + 1)) return;
  int iu = i / (GT_ + 1), iv = i % (GT_ + 1);
  float ae = (float)iu * (1.0f / GT_);
  float ap = (float)iv * (1.0f / GT_);
  Tg[i] = gate_exact(ae, ap, W1, b1, W2, b2);
}

__global__ void __launch_bounds__(256) build_bias_table(
    const float* __restrict__ W_bias, const float* __restrict__ b_bias,
    const float* __restrict__ bias_scale, float* __restrict__ Tb) {
  int i = blockIdx.x * 256 + threadIdx.x;
  if (i >= H_ * (BT_ + 1)) return;
  int h = i / (BT_ + 1), k = i % (BT_ + 1);
  float a = (float)k * (1.0f / BT_);
  Tb[i] = softplus_exact(a * W_bias[h] + b_bias[h]) * bias_scale[h];
}

// -------- gate MLP -> A_blended [B,N,N] f32 (bilinear table + exact fallback)
__global__ void __launch_bounds__(256) gate_kernel(
    const float* __restrict__ Ae, const float* __restrict__ Ap,
    const float* __restrict__ Tg,
    const float* __restrict__ W1, const float* __restrict__ b1,
    const float* __restrict__ W2, const float* __restrict__ b2,
    float* __restrict__ Ab, int n) {
  int i = blockIdx.x * 256 + threadIdx.x;
  if (i >= n) return;
  float ae = Ae[i], ap = Ap[i];
  float g;
  if (ae >= 0.0f && ae <= 1.0f && ap >= 0.0f && ap <= 1.0f) {
    float u = ae * (float)GT_, v = ap * (float)GT_;
    int iu = min((int)u, GT_ - 1), iv = min((int)v, GT_ - 1);
    float fu = u - (float)iu, fv = v - (float)iv;
    const float* t0 = Tg + (size_t)iu * (GT_ + 1) + iv;
    float g00 = t0[0], g01 = t0[1], g10 = t0[GT_ + 1], g11 = t0[GT_ + 2];
    float lo = g00 + (g01 - g00) * fv;
    float hi = g10 + (g11 - g10) * fv;
    g = lo + (hi - lo) * fu;
  } else {
    g = gate_exact(ae, ap, W1, b1, W2, b2);   // out-of-range fallback (exact)
  }
  Ab[i] = g * ae + (1.0f - g) * ap;
}

// --------------------- QKV GEMM (32x64 wave tiles): X[8192,512] x Wq^T -> bf16
__global__ void __launch_bounds__(256) qkv_gemm(
    const unsigned short* __restrict__ Xb, const unsigned short* __restrict__ Wq,
    const float* __restrict__ bias,
    unsigned short* __restrict__ Qb, unsigned short* __restrict__ Kb,
    unsigned short* __restrict__ Vb) {
  const int lane = threadIdx.x & 31;
  const int wid  = (blockIdx.x * 256 + threadIdx.x) >> 5;
  const int NT   = (3 * D_) / 64;                 // 24 col tiles
  const int rt = wid / NT, ct = wid % NT;
  const int row0 = rt * 32, col0 = ct * 64;
  const int g = lane >> 4, n16 = lane & 15;

  v8f acc[2][4];
#pragma unroll
  for (int rr = 0; rr < 2; ++rr)
#pragma unroll
    for (int i = 0; i < 4; ++i)
#pragma unroll
      for (int v = 0; v < 8; ++v) acc[rr][i][v] = 0.0f;

#pragma unroll 4
  for (int k = 0; k < D_; k += 32) {
    v16bf a0 = load_a_frag(Xb + (size_t)row0 * D_ + k, D_, lane);
    v16bf a1 = load_a_frag(Xb + (size_t)(row0 + 16) * D_ + k, D_, lane);
#pragma unroll
    for (int sub = 0; sub < 4; ++sub) {
      v16bf bf = load_b_frag(Wq + (size_t)(col0 + sub * 16) * D_ + k, D_, lane);
      acc[0][sub] = wmma_bf16(a0, bf, acc[0][sub]);
      acc[1][sub] = wmma_bf16(a1, bf, acc[1][sub]);
    }
  }
#pragma unroll
  for (int rr = 0; rr < 2; ++rr) {
#pragma unroll
    for (int sub = 0; sub < 4; ++sub) {
      int c = col0 + sub * 16 + n16;
      float bv = bias[c];
      int s = c / D_, rem = c % D_;
      int hh = rem / DH_, dd = rem % DH_;
      unsigned short* dst = (s == 0) ? Qb : (s == 1) ? Kb : Vb;
#pragma unroll
      for (int v = 0; v < 8; ++v) {
        int r = row0 + rr * 16 + v + 8 * g;
        int bb = r >> 10, nn = r & (N_ - 1);
        dst[(((size_t)bb * H_ + hh) * N_ + nn) * DH_ + dd] = f2bf(acc[rr][sub][v] + bv);
      }
    }
  }
}

// ----------------- fused flash attention with graph-bias, per (b,h,128 q-rows)
__global__ void __launch_bounds__(256) attn_kernel(
    const unsigned short* __restrict__ Qb, const unsigned short* __restrict__ Kb,
    const unsigned short* __restrict__ Vb, const float* __restrict__ Ab,
    const float* __restrict__ Tb,
    const float* __restrict__ W_bias, const float* __restrict__ b_bias,
    const float* __restrict__ bias_scale, unsigned short* __restrict__ Ob) {
  __shared__ __align__(16) unsigned short K2[2 * 32 * DH_]; // double-buffered K
#if ASYNC_LDS
  __shared__ __align__(16) unsigned short Vr[2 * 32 * DH_]; // double-buffered raw V
#endif
  __shared__ __align__(16) unsigned short VT[DH_ * 32];     // V chunk transposed
  __shared__ __align__(16) unsigned short Pl[8 * 16 * 32];  // per-wave P tiles
  __shared__ __align__(16) float TbS[BT_ + 1];              // this head's bias table

  const int tid = threadIdx.x, lane = tid & 31, wave = tid >> 5;
  const int bid = blockIdx.x;
  const int qb = bid & 7, bh = bid >> 3;          // 8 q-blocks of 128 rows
  const int b = bh >> 3, h = bh & 7;
  const int g = lane >> 4, n16 = lane & 15;
  const int NC = N_ / 32;

  const int q0 = qb * 128 + wave * 16;
  const unsigned short* Qbase = Qb + ((size_t)bh * N_ + q0) * DH_;
  const unsigned short* Kbh = Kb + (size_t)bh * N_ * DH_;
  const unsigned short* Vbh = Vb + (size_t)bh * N_ * DH_;
  const float* Abase = Ab + (size_t)b * N_ * N_;

  // stage this head's 1D bias table into LDS
  for (int i = tid; i < BT_ + 1; i += 256) TbS[i] = Tb[h * (BT_ + 1) + i];

  const float wb = W_bias[h], bb = b_bias[h], bs = bias_scale[h];
  const float scale = 0.125f;  // DH^-0.5

  const v16bf qf0 = load_a_frag(Qbase, DH_, lane);
  const v16bf qf1 = load_a_frag(Qbase + 32, DH_, lane);

  v8f o[4];
  float m[8], l[8];
#pragma unroll
  for (int i = 0; i < 4; ++i)
#pragma unroll
    for (int v = 0; v < 8; ++v) o[i][v] = 0.0f;
#pragma unroll
  for (int v = 0; v < 8; ++v) { m[v] = -1e30f; l[v] = 0.0f; }

  unsigned short* Pw = Pl + wave * 16 * 32;

#if ASYNC_LDS
  // prologue: async-fill chunk 0 into buffer 0 (2 async ops per wave)
  async_copy16(Kbh + tid * 8, K2 + tid * 8);
  async_copy16(Vbh + tid * 8, Vr + tid * 8);
#endif

  for (int jc = 0; jc < NC; ++jc) {
    const int cur = jc & 1, nxt = cur ^ 1;
    __syncthreads();  // previous readers of buffer `nxt` (and VT/P) are done

#if ASYNC_LDS
    {
      // issue fill of the *next* chunk (dummy refill on last iteration keeps
      // the wait count a compile-time constant; that buffer is never read)
      int jn = (jc + 1 < NC) ? (jc + 1) : (NC - 1);
      const unsigned short* sK = Kbh + (size_t)(jn * 32) * DH_;
      const unsigned short* sV = Vbh + (size_t)(jn * 32) * DH_;
      async_copy16(sK + tid * 8, K2 + nxt * (32 * DH_) + tid * 8);
      async_copy16(sV + tid * 8, Vr + nxt * (32 * DH_) + tid * 8);
    }
    // async loads complete in order: <=2 outstanding means `cur` group landed
    __builtin_amdgcn_s_wait_asynccnt(2);
    if (tid < 128) {  // prefetch next A_blended row segments
      int row = qb * 128 + tid;
      int jn = (jc + 1 < NC) ? (jc + 1) : (NC - 1);
      __builtin_prefetch(Abase + (size_t)row * N_ + jn * 32, 0, 1);
    }
#else
    {
      const unsigned short* srcK = Kbh + (size_t)(jc * 32) * DH_;
      const unsigned short* srcV = Vbh + (size_t)(jc * 32) * DH_;
      int i = tid;  // 256 threads move 256 x 16B = K tile + V tile (transposed)
      ((u32x4*)(K2 + cur * (32 * DH_)))[i] = ((const u32x4*)srcK)[i];
      union { u32x4 q; unsigned short u[8]; } w;
      w.q = ((const u32x4*)srcV)[i];
      int e = i * 8, kr = e >> 6, d0 = e & 63;
#pragma unroll
      for (int jj = 0; jj < 8; ++jj) VT[(d0 + jj) * 32 + kr] = w.u[jj];
      if (jc + 1 < NC) {
        if (tid < 32)       __builtin_prefetch(srcK + 32 * DH_ + tid * 64, 0, 1);
        else if (tid < 64)  __builtin_prefetch(srcV + 32 * DH_ + (tid - 32) * 64, 0, 1);
        else if (tid < 192) {
          int row = qb * 128 + (tid - 64);
          __builtin_prefetch(Abase + (size_t)row * N_ + (jc + 1) * 32, 0, 1);
        }
      }
    }
#endif
    __syncthreads();  // all waves' fills for `cur` visible

#if ASYNC_LDS
    {  // LDS->LDS transpose of V[cur] into VT (published by the P barrier below)
      union { u32x4 q; unsigned short u[8]; } w;
      w.q = ((const u32x4*)(Vr + cur * (32 * DH_)))[tid];
      int e = tid * 8, kr = e >> 6, d0 = e & 63;
#pragma unroll
      for (int jj = 0; jj < 8; ++jj) VT[(d0 + jj) * 32 + kr] = w.u[jj];
    }
#endif
    const unsigned short* Kcur = K2 + cur * (32 * DH_);

    // S = Q K^T (16x32 as two 16x16 subtiles; K-dim 64 = 2 WMMA each)
    v8f s[2];
#pragma unroll
    for (int t = 0; t < 2; ++t) {
#pragma unroll
      for (int v = 0; v < 8; ++v) s[t][v] = 0.0f;
      v16bf bf0 = load_b_frag(Kcur + t * 16 * DH_, DH_, lane);
      v16bf bf1 = load_b_frag(Kcur + t * 16 * DH_ + 32, DH_, lane);
      s[t] = wmma_bf16(qf0, bf0, s[t]);
      s[t] = wmma_bf16(qf1, bf1, s[t]);
    }

    // graph bias via LDS table (exact-softplus fallback out of range), clip
#pragma unroll
    for (int t = 0; t < 2; ++t) {
      int col = jc * 32 + t * 16 + n16;
#pragma unroll
      for (int v = 0; v < 8; ++v) {
        int row = q0 + v + 8 * g;
        float a = Abase[(size_t)row * N_ + col];
        float bias;
        if (a >= 0.0f && a <= 1.0f) {
          float u = a * (float)BT_;
          int iu = min((int)u, BT_ - 1);
          float fu = u - (float)iu;
          float t0 = TbS[iu], t1 = TbS[iu + 1];
          bias = t0 + (t1 - t0) * fu;
        } else {
          bias = softplus_exact(a * wb + bb) * bs;
        }
        float val = s[t][v] * scale + bias;
        s[t][v] = fminf(fmaxf(val, -50.0f), 50.0f);
      }
    }

    // online softmax (row reductions across 16-lane D-layout groups)
    float p0[8], p1[8];
#pragma unroll
    for (int v = 0; v < 8; ++v) {
      float t = fmaxf(s[0][v], s[1][v]);
#pragma unroll
      for (int off = 1; off < 16; off <<= 1) t = fmaxf(t, __shfl_xor(t, off, 16));
      float nm = fmaxf(m[v], t);
      float corr = __expf(m[v] - nm);
      p0[v] = __expf(s[0][v] - nm);
      p1[v] = __expf(s[1][v] - nm);
      float rs = p0[v] + p1[v];
#pragma unroll
      for (int off = 1; off < 16; off <<= 1) rs += __shfl_xor(rs, off, 16);
      l[v] = l[v] * corr + rs;
      m[v] = nm;
#pragma unroll
      for (int i = 0; i < 4; ++i) o[i][v] *= corr;
    }

    // D-layout -> A-fragment layout via LDS bounce
#pragma unroll
    for (int v = 0; v < 8; ++v) {
      int row = v + 8 * g;
      Pw[row * 32 + n16]      = f2bf(p0[v]);
      Pw[row * 32 + 16 + n16] = f2bf(p1[v]);
    }
    __syncthreads();   // publishes P tiles and VT

    v16bf pf = load_a_frag(Pw, 32, lane);
#pragma unroll
    for (int sub = 0; sub < 4; ++sub) {
      v16bf vf = load_b_frag(VT + sub * 16 * 32, 32, lane);
      o[sub] = wmma_bf16(pf, vf, o[sub]);
    }
  }

  // normalize and store per-head output (bf16) into [B,N,D] with col = h*64+d
#pragma unroll
  for (int sub = 0; sub < 4; ++sub) {
#pragma unroll
    for (int v = 0; v < 8; ++v) {
      int row = q0 + v + 8 * g;
      int col = h * DH_ + sub * 16 + n16;
      Ob[((size_t)b * N_ + row) * D_ + col] = f2bf(o[sub][v] / l[v]);
    }
  }
}

// ---------------- out proj GEMM (32x64 wave tiles): [8192,512]x[512,512] -> f32
__global__ void __launch_bounds__(256) proj_gemm(
    const unsigned short* __restrict__ Ain, const unsigned short* __restrict__ Wp,
    const float* __restrict__ bias, float* __restrict__ out) {
  const int lane = threadIdx.x & 31;
  const int wid  = (blockIdx.x * 256 + threadIdx.x) >> 5;
  const int NT   = D_ / 64;                       // 8 col tiles
  const int rt = wid / NT, ct = wid % NT;
  const int row0 = rt * 32, col0 = ct * 64;
  const int g = lane >> 4, n16 = lane & 15;

  v8f acc[2][4];
#pragma unroll
  for (int rr = 0; rr < 2; ++rr)
#pragma unroll
    for (int i = 0; i < 4; ++i)
#pragma unroll
      for (int v = 0; v < 8; ++v) acc[rr][i][v] = 0.0f;

#pragma unroll 4
  for (int k = 0; k < D_; k += 32) {
    v16bf a0 = load_a_frag(Ain + (size_t)row0 * D_ + k, D_, lane);
    v16bf a1 = load_a_frag(Ain + (size_t)(row0 + 16) * D_ + k, D_, lane);
#pragma unroll
    for (int sub = 0; sub < 4; ++sub) {
      v16bf bf = load_b_frag(Wp + (size_t)(col0 + sub * 16) * D_ + k, D_, lane);
      acc[0][sub] = wmma_bf16(a0, bf, acc[0][sub]);
      acc[1][sub] = wmma_bf16(a1, bf, acc[1][sub]);
    }
  }
#pragma unroll
  for (int rr = 0; rr < 2; ++rr) {
#pragma unroll
    for (int sub = 0; sub < 4; ++sub) {
      int c = col0 + sub * 16 + n16;
      float bv = bias[c];
#pragma unroll
      for (int v = 0; v < 8; ++v) {
        int r = row0 + rr * 16 + v + 8 * g;
        out[(size_t)r * D_ + c] = acc[rr][sub][v] + bv;
      }
    }
  }
}

extern "C" void kernel_launch(void* const* d_in, const int* in_sizes, int n_in,
                              void* d_out, int out_size, void* d_ws, size_t ws_size,
                              hipStream_t stream) {
  (void)in_sizes; (void)n_in; (void)out_size; (void)ws_size;
  const float* x      = (const float*)d_in[0];
  const float* Ae     = (const float*)d_in[1];
  const float* Ap     = (const float*)d_in[2];
  const float* W_qkv  = (const float*)d_in[3];
  const float* b_qkv  = (const float*)d_in[4];
  const float* W_proj = (const float*)d_in[5];
  const float* b_proj = (const float*)d_in[6];
  const float* W_g1   = (const float*)d_in[7];
  const float* b_g1   = (const float*)d_in[8];
  const float* W_g2   = (const float*)d_in[9];
  const float* b_g2   = (const float*)d_in[10];
  const float* W_bias = (const float*)d_in[11];
  const float* b_bias = (const float*)d_in[12];
  const float* bscale = (const float*)d_in[13];

  uint8_t* ws = (uint8_t*)d_ws;
  size_t off = 0;
  auto take = [&](size_t bytes) -> void* {
    void* p = ws + off;
    off += (bytes + 255) & ~(size_t)255;
    return p;
  };
  float*          Abl = (float*)take(sizeof(float) * (size_t)B_ * N_ * N_);   // 32 MB
  unsigned short* Xb  = (unsigned short*)take(2ull * B_ * N_ * D_);           //  8 MB
  unsigned short* Wqb = (unsigned short*)take(2ull * 3 * D_ * D_);
  unsigned short* Wpb = (unsigned short*)take(2ull * D_ * D_);
  unsigned short* Qb  = (unsigned short*)take(2ull * B_ * H_ * N_ * DH_);     //  8 MB
  unsigned short* Kb  = (unsigned short*)take(2ull * B_ * H_ * N_ * DH_);
  unsigned short* Vb  = (unsigned short*)take(2ull * B_ * H_ * N_ * DH_);
  unsigned short* Ob  = (unsigned short*)take(2ull * B_ * N_ * D_);           //  8 MB
  float*          Tg  = (float*)take(sizeof(float) * (GT_ + 1) * (GT_ + 1));  // ~1 MB
  float*          Tb  = (float*)take(sizeof(float) * H_ * (BT_ + 1));         // 33 KB

  const int nx = B_ * N_ * D_;
  cvt_kernel<<<(nx + 255) / 256, 256, 0, stream>>>(x, Xb, nx);
  cvt_kernel<<<(3 * D_ * D_ + 255) / 256, 256, 0, stream>>>(W_qkv, Wqb, 3 * D_ * D_);
  cvt_kernel<<<(D_ * D_ + 255) / 256, 256, 0, stream>>>(W_proj, Wpb, D_ * D_);

  build_gate_table<<<((GT_ + 1) * (GT_ + 1) + 255) / 256, 256, 0, stream>>>(
      W_g1, b_g1, W_g2, b_g2, Tg);
  build_bias_table<<<(H_ * (BT_ + 1) + 255) / 256, 256, 0, stream>>>(
      W_bias, b_bias, bscale, Tb);

  const int ng = B_ * N_ * N_;
  gate_kernel<<<ng / 256, 256, 0, stream>>>(Ae, Ap, Tg, W_g1, b_g1, W_g2, b_g2,
                                            Abl, ng);

  qkv_gemm<<<((B_ * N_ / 32) * (3 * D_ / 64)) / 8, 256, 0, stream>>>(
      Xb, Wqb, b_qkv, Qb, Kb, Vb);

  attn_kernel<<<B_ * H_ * (N_ / 128), 256, 0, stream>>>(
      Qb, Kb, Vb, Abl, Tb, W_bias, b_bias, bscale, Ob);

  proj_gemm<<<((B_ * N_ / 32) * (D_ / 64)) / 8, 256, 0, stream>>>(
      Ob, Wpb, b_proj, (float*)d_out);
}
